// ResidualVQ_38062000177392
// MI455X (gfx1250) — matrix operand order repaired and checked by reading, not compiled, not run on hardware
//
#include <hip/hip_runtime.h>

#define B_   16
#define T_   4096
#define DIN  512
#define DD   128
#define SZ   1024
#define NB   8
#define M_   (B_ * T_)   // 65536 rows

typedef __attribute__((ext_vector_type(16))) __bf16 v16bf;
typedef __attribute__((ext_vector_type(8)))  __bf16 bf16x8;
typedef __attribute__((ext_vector_type(2)))  __bf16 bf16x2;
typedef __attribute__((ext_vector_type(8)))  float  v8f;
typedef __attribute__((ext_vector_type(4)))  float  f32x4;
typedef __attribute__((ext_vector_type(4)))  unsigned int u32x4;

// ---------------- workspace layout (all offsets 256B aligned) ----------------
static const size_t OFF_RES   = 0;                                   // residual f32 [M,512]  134MB
static const size_t OFF_ZP    = OFF_RES   + (size_t)M_ * DIN * 4;    // z_proj f32 [M,128]    33.5MB
static const size_t OFF_ZPBF  = OFF_ZP    + (size_t)M_ * DD * 4;     // z_proj bf16           16.8MB
static const size_t OFF_CBBF  = OFF_ZPBF  + (size_t)M_ * DD * 2;     // codebooks bf16        2MB
static const size_t OFF_WIBF  = OFF_CBBF  + (size_t)NB * SZ * DD * 2;// W_in bf16             128KB
static const size_t OFF_WOBF  = OFF_WIBF  + (size_t)DD * DIN * 2;    // W_out bf16            128KB
static const size_t OFF_CNORM = OFF_WOBF  + (size_t)DIN * DD * 2;    // |c|^2 per code        32KB
static const size_t OFF_IDX   = OFF_CNORM + (size_t)NB * SZ * 4;     // idx int [M]           256KB
static const size_t OFF_LOSS  = OFF_IDX   + (size_t)M_ * 4;          // loss accumulator f32

// dynamic LDS (128KB in each GEMM kernel: 2 blocks co-resident per 320KB WGP)
extern __shared__ unsigned char dynsmem[];

// ---------------- bf16 conversion helpers ----------------
__device__ __forceinline__ __bf16 f2bf(float f) {
  unsigned u = __float_as_uint(f);
  unsigned r = (u + 0x7FFFu + ((u >> 16) & 1u)) >> 16;   // round-to-nearest-even
  unsigned short h = (unsigned short)r;
  __bf16 out;
  __builtin_memcpy(&out, &h, 2);
  return out;
}

#if defined(__has_builtin)
#if __has_builtin(__builtin_amdgcn_cvt_pk_bf16_f32)
#define HAVE_PK_BF16 1
#endif
#endif

__device__ __forceinline__ bf16x2 f2bf2(float x, float y) {
#ifdef HAVE_PK_BF16
  auto v = __builtin_amdgcn_cvt_pk_bf16_f32(x, y);
  bf16x2 r;
  constexpr unsigned n = sizeof(v) < sizeof(bf16x2) ? sizeof(v) : sizeof(bf16x2);
  __builtin_memcpy(&r, &v, n);
  return r;
#else
  bf16x2 r;
  r[0] = f2bf(x);
  r[1] = f2bf(y);
  return r;
#endif
}

__device__ __forceinline__ v8f zero8() {
  v8f z = {0.f, 0.f, 0.f, 0.f, 0.f, 0.f, 0.f, 0.f};
  return z;
}

// cooperative global->LDS copy of `bytes` (multiple of 16*blockDim) using 16B chunks
__device__ __forceinline__ void stage_to_lds(const void* gsrc, void* sdst, int n16) {
  const u32x4* s = (const u32x4*)gsrc;
  u32x4* d = (u32x4*)sdst;
  for (int i = threadIdx.x; i < n16; i += blockDim.x) d[i] = s[i];
}

// A-matrix 16x32 bf16 tile from a row-major bf16 matrix (lane<16: K 0-7 & 16-23)
__device__ __forceinline__ v16bf load_a_bf16(const __bf16* A, int lda, int m0, int k0, int lane) {
  int row = m0 + (lane & 15);
  int kb  = k0 + ((lane >> 4) << 3);
  const __bf16* p = A + (size_t)row * lda + kb;
  bf16x8 lo = *(const bf16x8*)(p);
  bf16x8 hi = *(const bf16x8*)(p + 16);
  v16bf a;
#pragma unroll
  for (int j = 0; j < 8; ++j) { a[j] = lo[j]; a[8 + j] = hi[j]; }
  return a;
}

// A-matrix 16x32 tile from a row-major f32 matrix, converting to bf16 on the fly
__device__ __forceinline__ v16bf load_a_f32(const float* A, int lda, int m0, int k0, int lane) {
  int row = m0 + (lane & 15);
  int kb  = k0 + ((lane >> 4) << 3);
  const float* p = A + (size_t)row * lda + kb;
  f32x4 q0 = *(const f32x4*)(p);
  f32x4 q1 = *(const f32x4*)(p + 4);
  f32x4 q2 = *(const f32x4*)(p + 16);
  f32x4 q3 = *(const f32x4*)(p + 20);
  v16bf a;
#pragma unroll
  for (int j = 0; j < 2; ++j) {
    bf16x2 p0 = f2bf2(q0[2 * j], q0[2 * j + 1]);
    bf16x2 p1 = f2bf2(q1[2 * j], q1[2 * j + 1]);
    bf16x2 p2 = f2bf2(q2[2 * j], q2[2 * j + 1]);
    bf16x2 p3 = f2bf2(q3[2 * j], q3[2 * j + 1]);
    a[2 * j]      = p0[0]; a[2 * j + 1]      = p0[1];
    a[4 + 2 * j]  = p1[0]; a[4 + 2 * j + 1]  = p1[1];
    a[8 + 2 * j]  = p2[0]; a[8 + 2 * j + 1]  = p2[1];
    a[12 + 2 * j] = p3[0]; a[12 + 2 * j + 1] = p3[1];
  }
  return a;
}

// B-matrix 32x16 bf16 tile; source stored N-major: Bt[n][k], ld = ldk.
// Lane<16: n=n0+lane, K=k0..k0+7 & k0+16..k0+23 ; lane>=16: same n, K=k0+8..k0+15 & k0+24..k0+31
__device__ __forceinline__ v16bf load_b_bf16(const __bf16* Bt, int ldk, int k0, int n0, int lane) {
  int n  = n0 + (lane & 15);
  int ks = k0 + ((lane >> 4) << 3);
  const __bf16* p = Bt + (size_t)n * ldk + ks;
  bf16x8 lo = *(const bf16x8*)(p);
  bf16x8 hi = *(const bf16x8*)(p + 16);
  v16bf b;
#pragma unroll
  for (int j = 0; j < 8; ++j) { b[j] = lo[j]; b[8 + j] = hi[j]; }
  return b;
}

__device__ __forceinline__ v8f wmma_bf16(v16bf a, v16bf b, v8f c) {
  return __builtin_amdgcn_wmma_f32_16x16x32_bf16(false, a, false, b, (short)0, c, false, false);
}

// ---------------- init: transpose z [B,512,T] -> residual [B*T,512], zero loss ----------------
__global__ void __launch_bounds__(256) k_tin(const float* __restrict__ z,
                                             float* __restrict__ res,
                                             float* __restrict__ loss) {
  __shared__ float tile[32][33];
  int tx = threadIdx.x, ty = threadIdx.y;
  int bx = blockIdx.x, by = blockIdx.y, bz = blockIdx.z;
#pragma unroll
  for (int j = 0; j < 4; ++j) {
    int Dc = by * 32 + ty + 8 * j;
    tile[ty + 8 * j][tx] = z[((size_t)bz * DIN + Dc) * T_ + bx * 32 + tx];
  }
  __syncthreads();
#pragma unroll
  for (int j = 0; j < 4; ++j) {
    int t = bx * 32 + ty + 8 * j;
    res[((size_t)bz * T_ + t) * DIN + by * 32 + tx] = tile[tx][ty + 8 * j];
  }
  if (bx == 0 && by == 0 && bz == 0 && tx == 0 && ty == 0) *loss = 0.f;
}

// ---------------- f32 -> bf16 conversion ----------------
__global__ void k_cvt(const float* __restrict__ src, __bf16* __restrict__ dst, int n) {
  int i = blockIdx.x * blockDim.x + threadIdx.x;
  if (2 * i + 1 < n) {
    bf16x2 p = f2bf2(src[2 * i], src[2 * i + 1]);
    *(bf16x2*)(dst + 2 * i) = p;
  } else if (2 * i < n) {
    dst[2 * i] = f2bf(src[2 * i]);
  }
}

// ---------------- per-code squared norms ----------------
__global__ void k_cnorm(const float* __restrict__ cb, float* __restrict__ cnorm, int n) {
  int s = blockIdx.x * blockDim.x + threadIdx.x;
  if (s >= n) return;
  const float* p = cb + (size_t)s * DD;
  float acc = 0.f;
#pragma unroll 8
  for (int d = 0; d < DD; ++d) acc += p[d] * p[d];
  cnorm[s] = acc;
}

// ---------------- book stage 1: z_proj = residual @ W_in^T + b_in ----------------
// W_in bf16 (128KB) staged in LDS once per block; all 8 waves share it.
__global__ void __launch_bounds__(256) k_proj(const float* __restrict__ res,
                                              const __bf16* __restrict__ wibf,
                                              const float* __restrict__ b_in,
                                              float* __restrict__ zp,
                                              __bf16* __restrict__ zpbf) {
  __bf16* sW = (__bf16*)dynsmem;                        // [128][512] bf16
  stage_to_lds(wibf, sW, DD * DIN * 2 / 16);
  __syncthreads();

  int lane = threadIdx.x & 31;
  int wid  = blockIdx.x * (blockDim.x >> 5) + (threadIdx.x >> 5);
  int m0   = wid * 16;
  v8f acc[8];
#pragma unroll
  for (int n = 0; n < 8; ++n) acc[n] = zero8();
  for (int kt = 0; kt < DIN / 32; ++kt) {
    v16bf a = load_a_f32(res, DIN, m0, kt * 32, lane);
#pragma unroll
    for (int nt = 0; nt < 8; ++nt) {
      v16bf b = load_b_bf16(sW, DIN, kt * 32, nt * 16, lane);   // ds_load
      acc[nt] = wmma_bf16(a, b, acc[nt]);
    }
  }
  int colb = lane & 15;
  int rb   = (lane >> 4) << 3;
#pragma unroll
  for (int nt = 0; nt < 8; ++nt) {
    int col = nt * 16 + colb;
    float bias = b_in[col];
#pragma unroll
    for (int i = 0; i < 8; ++i) {
      int row = m0 + rb + i;
      float v = acc[nt][i] + bias;
      zp[(size_t)row * DD + col]   = v;
      zpbf[(size_t)row * DD + col] = f2bf(v);
    }
  }
}

// ---------------- book stage 2: argmin_s ( |c_s|^2 - 2 z.c_s ) ----------------
// Codebook bf16 (256KB) staged in LDS in two 128KB halves (512 codes each).
__global__ void __launch_bounds__(256) k_argmin(const __bf16* __restrict__ zpbf,
                                                const __bf16* __restrict__ cbbf,   // this book
                                                const float*  __restrict__ cnorm,  // this book
                                                int*   __restrict__ idx_ws,
                                                float* __restrict__ codes,
                                                int book) {
  __bf16* scb = (__bf16*)dynsmem;                       // [512][128] bf16 per stage
  int lane = threadIdx.x & 31;
  int wid  = blockIdx.x * (blockDim.x >> 5) + (threadIdx.x >> 5);
  int m0   = wid * 16;
  v16bf a[4];
#pragma unroll
  for (int kt = 0; kt < 4; ++kt) a[kt] = load_a_bf16(zpbf, DD, m0, kt * 32, lane);
  float best[8];
  int   bidx[8];
#pragma unroll
  for (int i = 0; i < 8; ++i) { best[i] = 3.4e38f; bidx[i] = 0; }

  for (int stage = 0; stage < 2; ++stage) {
    stage_to_lds(cbbf + (size_t)stage * 512 * DD, scb, 512 * DD * 2 / 16);
    __syncthreads();
    for (int s0 = 0; s0 < 512; s0 += 32) {
      v8f acc0 = zero8(), acc1 = zero8();
#pragma unroll
      for (int kt = 0; kt < 4; ++kt) {
        v16bf b0 = load_b_bf16(scb, DD, kt * 32, s0, lane);        // ds_load
        acc0 = wmma_bf16(a[kt], b0, acc0);
      }
#pragma unroll
      for (int kt = 0; kt < 4; ++kt) {
        v16bf b1 = load_b_bf16(scb, DD, kt * 32, s0 + 16, lane);   // ds_load
        acc1 = wmma_bf16(a[kt], b1, acc1);
      }
      int c0 = stage * 512 + s0 + (lane & 15);
      int c1 = c0 + 16;
      float cn0 = cnorm[c0];
      float cn1 = cnorm[c1];
#pragma unroll
      for (int i = 0; i < 8; ++i) {
        float sc0 = cn0 - 2.f * acc0[i];
        if (sc0 < best[i]) { best[i] = sc0; bidx[i] = c0; }
        float sc1 = cn1 - 2.f * acc1[i];
        if (sc1 < best[i]) { best[i] = sc1; bidx[i] = c1; }
      }
    }
    __syncthreads();   // protect LDS before restage
  }

  // reduce over each 16-lane half (rows 0-7 in lanes 0-15, rows 8-15 in lanes 16-31)
#pragma unroll
  for (int msk = 1; msk <= 8; msk <<= 1) {
#pragma unroll
    for (int i = 0; i < 8; ++i) {
      float ov = __shfl_xor(best[i], msk, 32);
      int   oi = __shfl_xor(bidx[i], msk, 32);
      if (ov < best[i] || (ov == best[i] && oi < bidx[i])) { best[i] = ov; bidx[i] = oi; }
    }
  }
  if ((lane & 15) == 0) {
    int rb = (lane >> 4) << 3;
#pragma unroll
    for (int i = 0; i < 8; ++i) {
      int m = m0 + rb + i;
      idx_ws[m] = bidx[i];
      int b = m >> 12, t = m & (T_ - 1);
      codes[((size_t)b * NB + book) * T_ + t] = (float)bidx[i];
    }
  }
}

// ---------------- book stage 3: residual -= cb[idx] @ W_out^T + b_out ; loss += SSE ----------------
// W_out bf16 (128KB) staged in LDS once per block.
__global__ void __launch_bounds__(256) k_recon(const int*    __restrict__ idx_ws,
                                               const __bf16* __restrict__ cbbf,   // this book
                                               const float*  __restrict__ cb_f32, // this book
                                               const float*  __restrict__ zp,
                                               const __bf16* __restrict__ wobf,
                                               const float*  __restrict__ b_out,
                                               float* __restrict__ res,
                                               float* __restrict__ loss) {
  __bf16* sW = (__bf16*)dynsmem;                        // [512][128] bf16
  stage_to_lds(wobf, sW, DIN * DD * 2 / 16);
  __syncthreads();

  int lane = threadIdx.x & 31;
  int wid  = blockIdx.x * (blockDim.x >> 5) + (threadIdx.x >> 5);
  int m0   = wid * 16;
  int r    = lane & 15;
  int myidx = idx_ws[m0 + r];
  const __bf16* rowp = cbbf + (size_t)myidx * DD;

  // loss: lanes<16 -> d 0..63 of row r ; lanes>=16 -> d 64..127 of row r
  {
    const float* crow = cb_f32 + (size_t)myidx * DD;
    const float* zrow = zp + (size_t)(m0 + r) * DD;
    int d0 = (lane >> 4) * 64;
    float s = 0.f;
#pragma unroll 8
    for (int d = 0; d < 64; ++d) {
      float diff = crow[d0 + d] - zrow[d0 + d];
      s += diff * diff;
    }
#pragma unroll
    for (int msk = 1; msk <= 16; msk <<= 1) s += __shfl_xor(s, msk, 32);
    if (lane == 0) atomicAdd(loss, s);
  }

  // gathered A tiles: A row r = cb_bf16[idx[m0+r]]
  v16bf a[4];
#pragma unroll
  for (int kt = 0; kt < 4; ++kt) {
    int kb = kt * 32 + ((lane >> 4) << 3);
    bf16x8 lo = *(const bf16x8*)(rowp + kb);
    bf16x8 hi = *(const bf16x8*)(rowp + kb + 16);
#pragma unroll
    for (int j = 0; j < 8; ++j) { a[kt][j] = lo[j]; a[kt][8 + j] = hi[j]; }
  }

  int colb = lane & 15;
  int rb   = (lane >> 4) << 3;
  for (int nt = 0; nt < DIN / 16; nt += 2) {
    v8f acc0 = zero8(), acc1 = zero8();
#pragma unroll
    for (int kt = 0; kt < 4; ++kt) {
      v16bf b0 = load_b_bf16(sW, DD, kt * 32, nt * 16, lane);        // ds_load
      acc0 = wmma_bf16(a[kt], b0, acc0);
    }
#pragma unroll
    for (int kt = 0; kt < 4; ++kt) {
      v16bf b1 = load_b_bf16(sW, DD, kt * 32, (nt + 1) * 16, lane);  // ds_load
      acc1 = wmma_bf16(a[kt], b1, acc1);
    }
    int col0 = nt * 16 + colb;
    int col1 = col0 + 16;
    float bias0 = b_out[col0];
    float bias1 = b_out[col1];
#pragma unroll
    for (int i = 0; i < 8; ++i) {
      size_t o0 = (size_t)(m0 + rb + i) * DIN + col0;
      size_t o1 = (size_t)(m0 + rb + i) * DIN + col1;
      res[o0] = res[o0] - (acc0[i] + bias0);
      res[o1] = res[o1] - (acc1[i] + bias1);
    }
  }
}

// ---------------- finalize: z_quantized = z - residual (transpose back); write loss ----------------
__global__ void __launch_bounds__(256) k_fin(const float* __restrict__ z,
                                             const float* __restrict__ res,
                                             float* __restrict__ zq,
                                             const float* __restrict__ loss,
                                             float* __restrict__ loss_out) {
  __shared__ float tile[32][33];
  int tx = threadIdx.x, ty = threadIdx.y;
  int bx = blockIdx.x, by = blockIdx.y, bz = blockIdx.z;
#pragma unroll
  for (int j = 0; j < 4; ++j) {
    int t = bx * 32 + ty + 8 * j;
    tile[ty + 8 * j][tx] = res[((size_t)bz * T_ + t) * DIN + by * 32 + tx];
  }
  __syncthreads();
#pragma unroll
  for (int j = 0; j < 4; ++j) {
    int Dc = by * 32 + ty + 8 * j;
    size_t o = ((size_t)bz * DIN + Dc) * T_ + bx * 32 + tx;
    zq[o] = z[o] - tile[tx][ty + 8 * j];
  }
  if (bx == 0 && by == 0 && bz == 0 && tx == 0 && ty == 0)
    *loss_out = *loss * (1.25f / ((float)M_ * (float)DD));
}

// ---------------- launcher ----------------
extern "C" void kernel_launch(void* const* d_in, const int* in_sizes, int n_in,
                              void* d_out, int out_size, void* d_ws, size_t ws_size,
                              hipStream_t stream) {
  const float* z     = (const float*)d_in[0];
  const float* W_in  = (const float*)d_in[1];
  const float* b_in  = (const float*)d_in[2];
  const float* W_out = (const float*)d_in[3];
  const float* b_out = (const float*)d_in[4];
  const float* cb    = (const float*)d_in[5];

  char* ws = (char*)d_ws;
  float*  res   = (float*)(ws + OFF_RES);
  float*  zp    = (float*)(ws + OFF_ZP);
  __bf16* zpbf  = (__bf16*)(ws + OFF_ZPBF);
  __bf16* cbbf  = (__bf16*)(ws + OFF_CBBF);
  __bf16* wibf  = (__bf16*)(ws + OFF_WIBF);
  __bf16* wobf  = (__bf16*)(ws + OFF_WOBF);
  float*  cnorm = (float*)(ws + OFF_CNORM);
  int*    idxw  = (int*)(ws + OFF_IDX);
  float*  loss  = (float*)(ws + OFF_LOSS);

  float* zq       = (float*)d_out;
  float* codes    = zq + (size_t)B_ * DIN * T_;
  float* loss_out = codes + (size_t)B_ * NB * T_;

  dim3 tb(32, 8);
  dim3 tg(T_ / 32, DIN / 32, B_);
  k_tin<<<tg, tb, 0, stream>>>(z, res, loss);
  k_cvt<<<(DD * DIN / 2 + 255) / 256, 256, 0, stream>>>(W_in, wibf, DD * DIN);
  k_cvt<<<(DIN * DD / 2 + 255) / 256, 256, 0, stream>>>(W_out, wobf, DIN * DD);
  k_cvt<<<(NB * SZ * DD / 2 + 255) / 256, 256, 0, stream>>>(cb, cbbf, NB * SZ * DD);
  k_cnorm<<<(NB * SZ + 255) / 256, 256, 0, stream>>>(cb, cnorm, NB * SZ);

  const int gemm_blocks = (M_ / 16) / 8;      // 8 waves per block, one 16-row tile per wave
  const size_t LDS_BYTES = 128 * 1024;        // 128KB -> 2 blocks per 320KB WGP
  for (int book = 0; book < NB; ++book) {
    k_proj<<<gemm_blocks, 256, LDS_BYTES, stream>>>(res, wibf, b_in, zp, zpbf);
    k_argmin<<<gemm_blocks, 256, LDS_BYTES, stream>>>(zpbf, cbbf + (size_t)book * SZ * DD,
                                                      cnorm + book * SZ, idxw, codes, book);
    k_recon<<<gemm_blocks, 256, LDS_BYTES, stream>>>(idxw, cbbf + (size_t)book * SZ * DD,
                                                     cb + (size_t)book * SZ * DD, zp, wobf,
                                                     b_out, res, loss);
  }
  k_fin<<<tg, tb, 0, stream>>>(z, res, zq, loss, loss_out);
}
